// VGN_74560632258880
// MI455X (gfx1250) — matrix operand order repaired and checked by reading, not compiled, not run on hardware
//
#include <hip/hip_runtime.h>
#include <hip/hip_bf16.h>

typedef __attribute__((ext_vector_type(16))) _Float16 v16h;
typedef __attribute__((ext_vector_type(8)))  float    v8f;

#define N_AG   64
#define OBSD   128
#define LDS_ST 138   // halves per row; 276B stride = 69 dwords (gcd(5,64)=1) -> conflict-free

// ---------------------------------------------------------------------------
// Prologue: w1 = W @ a[:H], w2 = W @ a[H:], stored f16 into workspace.
// W is [OBS=128, H=64] row-major, a is [128,1].
// ---------------------------------------------------------------------------
__global__ void prep_wv(const float* __restrict__ W, const float* __restrict__ a,
                        _Float16* __restrict__ wv) {
  const int k = threadIdx.x;           // 0..127
  float s1 = 0.f, s2 = 0.f;
#pragma unroll 8
  for (int h = 0; h < 64; ++h) {
    const float w = W[k * 64 + h];
    s1 += w * a[h];
    s2 += w * a[64 + h];
  }
  wv[k]       = (_Float16)s1;
  wv[128 + k] = (_Float16)s2;
}

// ---------------------------------------------------------------------------
// Fused GAT-mixing kernel: one workgroup per batch element.
// ---------------------------------------------------------------------------
__global__ __launch_bounds__(256) void gat_qtot(
    const float* __restrict__ obs,   // [B, 64*128]
    const float* __restrict__ qs,    // [B, 64]
    const int*   __restrict__ adj,   // [B, 64, 64]
    const _Float16* __restrict__ wv, // [2*128] f16 (w1 | w2)
    float* __restrict__ out)         // [B]
{
  __shared__ _Float16 hobs[N_AG * LDS_ST];   // obs tile, f16, padded rows
  __shared__ _Float16 wshB[16 * LDS_ST];     // dense zero-padded B (cols 0/1 = w1/w2)
  __shared__ float s_src[N_AG];
  __shared__ float s_dst[N_AG];
  __shared__ float red[N_AG];

  const int b   = blockIdx.x;
  const int tid = threadIdx.x;

  // stage dense zero-padded B matrix: wshB[n][k] = (n==0? w1[k] : n==1? w2[k] : 0)
  for (int i = tid; i < 16 * LDS_ST; i += 256) {
    const int n = i / LDS_ST;
    const int k = i - n * LDS_ST;
    _Float16 val = (_Float16)0.f;
    if (n < 2 && k < OBSD) val = wv[n * OBSD + k];
    wshB[i] = val;
  }

  // stage obs[b] -> LDS as f16 (coalesced float4 loads, 8 per thread)
  const float4* obs4 = (const float4*)(obs + (size_t)b * (N_AG * OBSD));
#pragma unroll
  for (int it = 0; it < 8; ++it) {
    const int idx = tid + it * 256;          // 0..2047 float4 index
    const float4 x = obs4[idx];
    const int r = idx >> 5;                  // 32 float4 per row
    const int c = (idx & 31) << 2;
    _Float16* dst = &hobs[r * LDS_ST + c];
    dst[0] = (_Float16)x.x; dst[1] = (_Float16)x.y;
    dst[2] = (_Float16)x.z; dst[3] = (_Float16)x.w;
  }
  __syncthreads();

  // ---- Phase 1: S(64x16) = obs_f16(64x128) @ [w1 w2 0..](128x16) via WMMA ----
  const int wave = tid >> 5;                 // wave32
  const int lane = tid & 31;
  if (wave < 4) {                            // wave w owns rows 16w..16w+15
    const int mrow = (wave << 4) + (lane & 15);
    const int hi   = lane >> 4;              // lane half selects K sub-range
    const int n    = lane & 15;              // output column owned by lane
    const _Float16* arow = &hobs[mrow * LDS_ST];
    const _Float16* bcol = &wshB[n * LDS_ST];
    v8f acc = {0.f, 0.f, 0.f, 0.f, 0.f, 0.f, 0.f, 0.f};
#pragma unroll
    for (int kc = 0; kc < 4; ++kc) {         // K = 128 in 4 chunks of 32
      // A fragment: 16-bit A 16x32 layout (ISA 7.12.2); all offsets even -> b32 pairs
      v16h af;
#pragma unroll
      for (int vv = 0; vv < 8; ++vv) {
        const int kk = kc * 32 + ((vv < 4) ? (hi * 8 + 2 * vv)
                                           : (16 + hi * 8 + 2 * (vv - 4)));
        af[2 * vv]     = arow[kk];
        af[2 * vv + 1] = arow[kk + 1];
      }
      // B fragment: 16-bit B 32x16 layout; unconditional loads from dense tile
      v16h bf;
#pragma unroll
      for (int vv = 0; vv < 8; ++vv) {
        const int k0 = kc * 32 + ((vv < 4) ? (hi * 8 + 2 * vv)
                                           : (16 + hi * 8 + 2 * (vv - 4)));
        bf[2 * vv]     = bcol[k0];
        bf[2 * vv + 1] = bcol[k0 + 1];
      }
      acc = __builtin_amdgcn_wmma_f32_16x16x32_f16(
          /*neg_a=*/false, af, /*neg_b=*/false, bf,
          /*c_mod=*/(short)0, acc, /*reuse_a=*/false, /*reuse_b=*/false);
    }
    // C layout: lanes 0-15 hold N=lane, M=r; lanes 16-31 hold N=lane-16, M=8+r
    if (n < 2) {
      float* dstv = (n == 0) ? s_src : s_dst;
      const int mb = (wave << 4) + hi * 8;
#pragma unroll
      for (int r = 0; r < 8; ++r) dstv[mb + r] = acc[r];
    }
  }
  __syncthreads();

  // ---- Phase 2: masked leaky-relu scores, softmax over i, weighted sum ----
  float contrib = 0.f;
  if (tid < N_AG) {
    const int j = tid;
    const float sdj = s_dst[j];
    const int* adjb = adj + (size_t)b * (N_AG * N_AG) + j;   // column j, stride 64

    unsigned long long mask = 0ull;                          // adj column bitmask
#pragma unroll 4
    for (int i = 0; i < N_AG; ++i)
      if (adjb[i * N_AG] > 0) mask |= (1ull << i);

    float mx = -1.0e12f;
#pragma unroll 4
    for (int i = 0; i < N_AG; ++i) {
      const float s = s_src[i] + sdj;
      const float e = (s >= 0.f) ? s : 0.2f * s;
      const float v = ((mask >> i) & 1ull) ? e : -1.0e12f;
      mx = fmaxf(mx, v);
    }
    float sum = 0.f;
#pragma unroll 4
    for (int i = 0; i < N_AG; ++i) {
      const float s = s_src[i] + sdj;
      const float e = (s >= 0.f) ? s : 0.2f * s;
      const float v = ((mask >> i) & 1ull) ? e : -1.0e12f;
      sum += __expf(v - mx);
    }
    const float inv = 1.f / sum;
    float attn_qs = 0.f;
#pragma unroll 4
    for (int i = 0; i < N_AG; ++i) {
      const float s = s_src[i] + sdj;
      const float e = (s >= 0.f) ? s : 0.2f * s;
      const float v = ((mask >> i) & 1ull) ? e : -1.0e12f;
      attn_qs += __expf(v - mx) * inv;
    }
    contrib = attn_qs * qs[(size_t)b * N_AG + j];
  }
  if (tid < N_AG) red[tid] = contrib;
  __syncthreads();
  if (tid == 0) {
    float t = 0.f;
#pragma unroll 8
    for (int i = 0; i < N_AG; ++i) t += red[i];
    out[b] = t;
  }
}

// ---------------------------------------------------------------------------
extern "C" void kernel_launch(void* const* d_in, const int* in_sizes, int n_in,
                              void* d_out, int out_size, void* d_ws, size_t ws_size,
                              hipStream_t stream) {
  const float* obs = (const float*)d_in[0];   // agent_obses [B, 64*128]
  const float* q   = (const float*)d_in[1];   // agent_qs    [B, 64]
  const int*   adj = (const int*)  d_in[2];   // adj         [B, 64, 64]
  const float* W   = (const float*)d_in[3];   // W           [128, 64]
  const float* a   = (const float*)d_in[4];   // a           [128, 1]
  float* out = (float*)d_out;
  _Float16* wv = (_Float16*)d_ws;             // 256 halves = 512 B of workspace

  const int B = in_sizes[0] / (N_AG * OBSD);

  prep_wv<<<1, 128, 0, stream>>>(W, a, wv);
  gat_qtot<<<B, 256, 0, stream>>>(obs, q, adj, wv, out);
}
